// MAEViT2D_52501680226425
// MI455X (gfx1250) — matrix operand order repaired and checked by reading, not compiled, not run on hardware
//
#include <hip/hip_runtime.h>
#include <hip/hip_bf16.h>
#include <cstdint>
#include <cstddef>

// ---------------------------------------------------------------------------
// Types for CDNA5 WMMA (wave32): D(f32 16x16) = A(bf16 16x32) x B(bf16 32x16) + C
// ---------------------------------------------------------------------------
typedef __attribute__((ext_vector_type(16))) __bf16 bf16x16;
typedef __attribute__((ext_vector_type(8)))  __bf16 bf16x8;
typedef __attribute__((ext_vector_type(8)))  float  f32x8;

#define WMMA_BF16(a, b, c) \
  __builtin_amdgcn_wmma_f32_16x16x32_bf16(false, (a), false, (b), (short)0, (c), false, false)

static __device__ __forceinline__ f32x8 zero_f32x8() {
  f32x8 z;
#pragma unroll
  for (int i = 0; i < 8; ++i) z[i] = 0.0f;
  return z;
}

// ---------------------------------------------------------------------------
// CDNA5 async global->LDS copy (ISA 15.18.3, ASYNCcnt) via inline asm.
// Per-lane: LDS[ldsoff] <- 16 bytes at gsrc. Generic LDS pointer low 32 bits
// are the LDS byte offset (ISA 10.2 aperture mapping).
// Async loads of one wave retire IN ORDER (ISA 4.1), so waiting
// asynccnt <= N keeps the newest N ops in flight while older ones are done.
// ---------------------------------------------------------------------------
static __device__ __forceinline__ void async_g2l_b128(const __bf16* lds_dst,
                                                      const __bf16* gsrc) {
  uint32_t loff = (uint32_t)(size_t)(const void*)lds_dst;
  uint64_t ga   = (uint64_t)(size_t)(const void*)gsrc;
  asm volatile("global_load_async_to_lds_b128 %0, %1, off"
               :: "v"(loff), "v"(ga) : "memory");
}
static __device__ __forceinline__ void wait_async_0() {
  asm volatile("s_wait_asynccnt 0x0" ::: "memory");
}
static __device__ __forceinline__ void wait_async_4() {
  asm volatile("s_wait_asynccnt 0x4" ::: "memory");
}

// A-matrix fragment (16x32 bf16), rows of length >=32 at stride ld (elements).
// Layout: lane<16 -> K 0..15 ; lane>=16 -> K 16..31, M = lane&15.
static __device__ __forceinline__ bf16x16 fragA(const __bf16* base, int ld) {
  const int lane = threadIdx.x & 31;
  const int m  = lane & 15;
  const int kb = (lane >> 4) << 4;              // 0 or 16
  const __bf16* p = base + (size_t)m * ld + kb;
  bf16x8 lo = *(const bf16x8*)(p);
  bf16x8 hi = *(const bf16x8*)(p + 8);
  bf16x16 f;
#pragma unroll
  for (int i = 0; i < 8; ++i) { f[i] = lo[i]; f[8 + i] = hi[i]; }
  return f;
}

// B-matrix fragment (32x16) where B[k][n] is stored as rows n of length >=32
// (i.e. B = W^T with W row-major N x K). lane<16 -> K 0..15, lane>=16 -> K 16..31.
static __device__ __forceinline__ bf16x16 fragB_rows(const __bf16* base, int ld) {
  const int lane = threadIdx.x & 31;
  const int n  = lane & 15;
  const int kb = (lane >> 4) << 4;              // 0 or 16
  const __bf16* p = base + (size_t)n * ld + kb;
  bf16x8 lo = *(const bf16x8*)(p);
  bf16x8 hi = *(const bf16x8*)(p + 8);
  bf16x16 f;
#pragma unroll
  for (int i = 0; i < 8; ++i) { f[i] = lo[i]; f[8 + i] = hi[i]; }
  return f;
}

// B-matrix fragment (32x16) stored truly column-major-ish: B[k][n] at base[k*ld + n].
static __device__ __forceinline__ bf16x16 fragB_cols(const __bf16* base, int ld) {
  const int lane = threadIdx.x & 31;
  const int n  = lane & 15;
  const int kb = (lane >> 4) << 4;              // 0 or 16
  bf16x16 f;
#pragma unroll
  for (int i = 0; i < 16; ++i) f[i] = base[(size_t)(kb + i) * ld + n];
  return f;
}

// ---------------------------------------------------------------------------
// GEMM: Y[M,N] = X[M,K](bf16) * W[N,K]^T(bf16) + bias (+res) (opt GELU)
// Block 256 = 8 waves in 4(M)x2(N); block tile 128x128; wave tile 32x64
// (2 A-frags x 4 B-frags = 8 WMMA per K-step). M % 128 == 0, K % 32 == 0.
// Fast path: DOUBLE-BUFFERED async global->LDS staging (ASYNCcnt pipeline).
// Edge path (N tile partial, heads N=64): guarded register staging.
// ---------------------------------------------------------------------------
__global__ __launch_bounds__(256) void gemm_bf16_kernel(
    const __bf16* __restrict__ X, const __bf16* __restrict__ W,
    const float* __restrict__ bias, const float* __restrict__ res,
    float* __restrict__ Yf, __bf16* __restrict__ Yb,
    int M, int N, int K, int act_gelu)
{
  __shared__ __bf16 Xs[2][128 * 40];   // double buffer: 128 rows, 32 cols + pad 8
  __shared__ __bf16 Ws[2][128 * 40];
  const int tid  = threadIdx.x;
  const int lane = tid & 31;
  const int wave = tid >> 5;
  const int wm = wave & 3;          // 0..3 -> M sub-tile
  const int wn = wave >> 2;         // 0..1 -> N sub-tile
  const int m0 = blockIdx.y * 128;
  const int n0 = blockIdx.x * 128;

  f32x8 acc[2][4];
#pragma unroll
  for (int i = 0; i < 2; ++i)
#pragma unroll
    for (int j = 0; j < 4; ++j) acc[i][j] = zero_f32x8();

  if (n0 + 128 <= N) {
    // ------- fast path: double-buffered async global->LDS pipeline -------
    auto issue = [&](int buf, int k0) {
#pragma unroll
      for (int j = 0; j < 2; ++j) {
        const int id  = tid + j * 256;          // 512 v8 slots per tile
        const int row = id >> 2, seg = id & 3;
        async_g2l_b128(&Xs[buf][row * 40 + seg * 8],
                       X + (size_t)(m0 + row) * K + k0 + seg * 8);
        async_g2l_b128(&Ws[buf][row * 40 + seg * 8],
                       W + (size_t)(n0 + row) * K + k0 + seg * 8);
      }
    };

    const int nsteps = K >> 5;
    issue(0, 0);
    for (int s = 0; s < nsteps; ++s) {
      const int cur = s & 1;
      const bool more = (s + 1 < nsteps);
      if (more) {
        issue(cur ^ 1, (s + 1) << 5);   // prefetch next K-step into other buffer
        wait_async_4();                 // 4 newest in flight => batch s landed
      } else {
        wait_async_0();
      }
      __syncthreads();

      bf16x16 a0 = fragA(&Xs[cur][(wm * 32 + 0)  * 40], 40);
      bf16x16 a1 = fragA(&Xs[cur][(wm * 32 + 16) * 40], 40);
#pragma unroll
      for (int tj = 0; tj < 4; ++tj) {
        bf16x16 b = fragB_rows(&Ws[cur][(wn * 64 + tj * 16) * 40], 40);
        acc[0][tj] = WMMA_BF16(a0, b, acc[0][tj]);
        acc[1][tj] = WMMA_BF16(a1, b, acc[1][tj]);
      }
      __syncthreads();   // all waves done reading buf cur; safe to overwrite
    }
  } else {
    // ---------------- edge path: guarded register staging ----------------
    for (int k0 = 0; k0 < K; k0 += 32) {
#pragma unroll
      for (int j = 0; j < 2; ++j) {
        const int id  = tid + j * 256;
        const int row = id >> 2, seg = id & 3;
        *(bf16x8*)&Xs[0][row * 40 + seg * 8] =
            *(const bf16x8*)(X + (size_t)(m0 + row) * K + k0 + seg * 8);
        bf16x8 v;
#pragma unroll
        for (int e = 0; e < 8; ++e) v[e] = (__bf16)0.0f;
        if (n0 + row < N)
          v = *(const bf16x8*)(W + (size_t)(n0 + row) * K + k0 + seg * 8);
        *(bf16x8*)&Ws[0][row * 40 + seg * 8] = v;
      }
      __syncthreads();

      bf16x16 a0 = fragA(&Xs[0][(wm * 32 + 0)  * 40], 40);
      bf16x16 a1 = fragA(&Xs[0][(wm * 32 + 16) * 40], 40);
#pragma unroll
      for (int tj = 0; tj < 4; ++tj) {
        bf16x16 b = fragB_rows(&Ws[0][(wn * 64 + tj * 16) * 40], 40);
        acc[0][tj] = WMMA_BF16(a0, b, acc[0][tj]);
        acc[1][tj] = WMMA_BF16(a1, b, acc[1][tj]);
      }
      __syncthreads();
    }
  }

  const int mm = (lane >> 4) << 3;   // 0 or 8
  const int nn = lane & 15;
#pragma unroll
  for (int ti = 0; ti < 2; ++ti)
#pragma unroll
    for (int tj = 0; tj < 4; ++tj) {
      const int n = n0 + wn * 64 + tj * 16 + nn;
      if (n >= N) continue;
      const float bv = bias ? bias[n] : 0.0f;
#pragma unroll
      for (int r = 0; r < 8; ++r) {
        const int m = m0 + wm * 32 + ti * 16 + mm + r;
        const size_t off = (size_t)m * N + n;
        float v = acc[ti][tj][r] + bv;
        if (res) v += res[off];
        if (act_gelu) v = 0.5f * v * (1.0f + erff(v * 0.70710678118654752f));
        if (Yf) Yf[off] = v;
        if (Yb) Yb[off] = (__bf16)v;
      }
    }
}

// ---------------------------------------------------------------------------
// Fused attention per (b, h, 16-query tile). Block 256 = 8 waves.
// qkv: bf16 (B, Lx, 3*C) with q|k|v at offsets 0|C|2C, head h at h*64. hd=64.
// Dynamic LDS: Sf(16*Lx f32) | Pb(16*Lx bf16) | Vs(Lx*72 bf16) | Qs(16*72) | rinv | qpos
// ---------------------------------------------------------------------------
__global__ __launch_bounds__(256) void attn_kernel(
    const __bf16* __restrict__ qkv, const float* __restrict__ bt,
    const int* __restrict__ vis, __bf16* __restrict__ O,
    int Lx, int C, int H)
{
  extern __shared__ char smem[];
  float*  Sf   = (float*)smem;                                   // 16*Lx
  __bf16* Pb   = (__bf16*)(smem + (size_t)16 * Lx * 4);          // 16*Lx
  __bf16* Vs   = Pb + (size_t)16 * Lx;                           // Lx*72
  __bf16* Qs   = Vs + (size_t)Lx * 72;                           // 16*72
  float*  rinv = (float*)(Qs + 16 * 72);                         // 16
  int*    qpos = (int*)(rinv + 16);                              // 16
  float*  Opart = Sf;                                            // reuse after softmax

  const int tid  = threadIdx.x;
  const int lane = tid & 31;
  const int wave = tid >> 5;
  const int l0 = blockIdx.x << 4;
  const int h  = blockIdx.y;
  const int b  = blockIdx.z;
  const int rs3 = 3 * C;
  const __bf16* base = qkv + (size_t)b * Lx * rs3;

  // ---- phase 0: async-stage Q tile + full V panel, query positions ----
  if (tid < 16) { const int lq = l0 + tid; qpos[tid] = vis ? vis[lq] : lq; }
  for (int id = tid; id < 16 * 8; id += 256) {
    const int r = id >> 3, sgi = id & 7;
    async_g2l_b128(&Qs[r * 72 + sgi * 8],
                   base + (size_t)(l0 + r) * rs3 + h * 64 + sgi * 8);
  }
  for (int id = tid; id < Lx * 8; id += 256) {
    const int r = id >> 3, sgi = id & 7;
    async_g2l_b128(&Vs[r * 72 + sgi * 8],
                   base + (size_t)r * rs3 + 2 * C + h * 64 + sgi * 8);
  }
  wait_async_0();
  __syncthreads();

  // ---- phase 1: S = (Q K^T) * 1/8 + rel-pos bias ----
  const int nkt = Lx >> 4;
  for (int kt = wave; kt < nkt; kt += 8) {
    const int k0 = kt << 4;
    const __bf16* kptr = base + (size_t)k0 * rs3 + C + h * 64;
    f32x8 acc = zero_f32x8();
    bf16x16 a0 = fragA(Qs, 72);
    bf16x16 b0 = fragB_rows(kptr, rs3);
    acc = WMMA_BF16(a0, b0, acc);
    bf16x16 a1 = fragA(Qs + 32, 72);
    bf16x16 b1 = fragB_rows(kptr + 32, rs3);
    acc = WMMA_BF16(a1, b1, acc);

    const int kq = lane & 15;
    const int lk = k0 + kq;
    const int pk = vis ? vis[lk] : lk;
    const int kh = pk >> 5, kw = pk & 31;
    const int mm = (lane >> 4) << 3;
#pragma unroll
    for (int r = 0; r < 8; ++r) {
      const int q  = mm + r;
      const int pq = qpos[q];
      const int relidx = ((pq >> 5) - kh + 31) * 63 + ((pq & 31) - kw + 31);
      Sf[q * Lx + lk] = acc[r] * 0.125f + bt[relidx * H + h];
    }
  }
  __syncthreads();

  // ---- phase 2: row softmax (unnormalized exp in bf16, keep 1/sum) ----
  {
    const int q = tid >> 4, sub = tid & 15;
    float* Srow  = Sf + q * Lx;
    __bf16* Prow = Pb + q * Lx;
    float mx = -3.0e38f;
    for (int k = sub; k < Lx; k += 16) mx = fmaxf(mx, Srow[k]);
#pragma unroll
    for (int o = 8; o >= 1; o >>= 1) mx = fmaxf(mx, __shfl_xor(mx, o, 16));
    float sum = 0.0f;
    for (int k = sub; k < Lx; k += 16) {
      const float e = __expf(Srow[k] - mx);
      sum += e;
      Prow[k] = (__bf16)e;
    }
#pragma unroll
    for (int o = 8; o >= 1; o >>= 1) sum += __shfl_xor(sum, o, 16);
    if (sub == 0) rinv[q] = 1.0f / sum;
  }
  __syncthreads();

  // ---- phase 3: O = P V  (4 d-tiles x 2 k-halves over 8 waves) ----
  {
    const int ntile = wave & 3;
    const int khalf = wave >> 2;
    const int kbeg = khalf * (Lx >> 1);
    const int kend = kbeg + (Lx >> 1);
    f32x8 acc = zero_f32x8();
    for (int kk = kbeg; kk < kend; kk += 32) {
      bf16x16 a  = fragA(Pb + kk, Lx);
      bf16x16 bv = fragB_cols(Vs + (size_t)kk * 72 + ntile * 16, 72);
      acc = WMMA_BF16(a, bv, acc);
    }
    const int d  = ntile * 16 + (lane & 15);
    const int mm = (lane >> 4) << 3;
#pragma unroll
    for (int r = 0; r < 8; ++r)
      Opart[(khalf * 16 + mm + r) * 64 + d] = acc[r];
  }
  __syncthreads();
  for (int e = tid; e < 1024; e += 256) {
    const int q = e >> 6, d = e & 63;
    const float v = (Opart[q * 64 + d] + Opart[(16 + q) * 64 + d]) * rinv[q];
    O[((size_t)b * Lx + l0 + q) * C + h * 64 + d] = (__bf16)v;
  }
}

// ---------------------------------------------------------------------------
// LayerNorm (population var) -> bf16 output. One row per block (256 threads).
// ---------------------------------------------------------------------------
__global__ __launch_bounds__(256) void layernorm_bf16_kernel(
    const float* __restrict__ x, const float* __restrict__ g,
    const float* __restrict__ b, __bf16* __restrict__ y, int C)
{
  const int row = blockIdx.x;
  const int tid = threadIdx.x;
  const float* xr = x + (size_t)row * C;
  float s = 0.0f, s2 = 0.0f;
  for (int i = tid; i < C; i += 256) { const float v = xr[i]; s += v; s2 += v * v; }
#pragma unroll
  for (int o = 16; o >= 1; o >>= 1) {
    s  += __shfl_xor(s,  o, 32);
    s2 += __shfl_xor(s2, o, 32);
  }
  __shared__ float red[16];
  __shared__ float mv[2];
  const int lane = tid & 31, wave = tid >> 5;
  if (lane == 0) { red[wave] = s; red[8 + wave] = s2; }
  __syncthreads();
  if (tid == 0) {
    float a = 0.0f, a2 = 0.0f;
    for (int w = 0; w < 8; ++w) { a += red[w]; a2 += red[8 + w]; }
    const float mean = a / C;
    mv[0] = mean;
    mv[1] = rsqrtf(a2 / C - mean * mean + 1e-5f);
  }
  __syncthreads();
  const float mean = mv[0], rstd = mv[1];
  __bf16* yr = y + (size_t)row * C;
  for (int i = tid; i < C; i += 256)
    yr[i] = (__bf16)((xr[i] - mean) * rstd * g[i] + b[i]);
}

// ---------------------------------------------------------------------------
// Elementwise helpers
// ---------------------------------------------------------------------------
__global__ void f32_to_bf16_kernel(const float* __restrict__ s, __bf16* __restrict__ d, long n) {
  const long i = (long)blockIdx.x * 256 + threadIdx.x;
  if (i < n) d[i] = (__bf16)s[i];
}

// x (8,2,256,256) -> tokens bf16 (8,1024,128)
__global__ void patchify_kernel(const float* __restrict__ x, __bf16* __restrict__ tok) {
  const int idx = blockIdx.x * 256 + threadIdx.x;   // 1,048,576
  const int t = idx & 127;
  const int l = (idx >> 7) & 1023;
  const int b = idx >> 17;
  const int c  = t & 1;
  const int pc = (t >> 1) & 7;
  const int pr = t >> 4;
  const int ih = (l >> 5) * 8 + pr, iw = (l & 31) * 8 + pc;
  tok[idx] = (__bf16)x[(((size_t)b * 2 + c) * 256 + ih) * 256 + iw];
}

__global__ void gather_tokens_kernel(const __bf16* __restrict__ tok,
                                     const int* __restrict__ vis, __bf16* __restrict__ tv) {
  const int idx = blockIdx.x * 256 + threadIdx.x;   // 524,288
  const int k = idx & 127;
  const int i = (idx >> 7) & 511;
  const int b = idx >> 16;
  tv[idx] = tok[((size_t)b * 1024 + vis[i]) * 128 + k];
}

__global__ void add_pos_vis_kernel(float* __restrict__ X, const float* __restrict__ pos,
                                   const int* __restrict__ vis) {
  const int idx = blockIdx.x * 256 + threadIdx.x;   // 3,145,728
  const int c = idx % 768;
  const int i = (idx / 768) & 511;
  X[idx] += pos[(size_t)vis[i] * 768 + c];
}

__global__ void posinvis_init_kernel(int* p) {
  const int i = blockIdx.x * 256 + threadIdx.x;
  if (i < 1024) p[i] = -1;
}
__global__ void posinvis_set_kernel(const int* __restrict__ vis, int* __restrict__ p) {
  const int i = blockIdx.x * 256 + threadIdx.x;
  if (i < 512) p[vis[i]] = i;
}

__global__ void scatter_dec_kernel(float* __restrict__ Xd, const float* __restrict__ dv,
                                   const float* __restrict__ mtok, const float* __restrict__ posd,
                                   const int* __restrict__ piv) {
  const int idx = blockIdx.x * 256 + threadIdx.x;   // 4,194,304
  const int c = idx & 511;
  const int l = (idx >> 9) & 1023;
  const int b = idx >> 19;
  const int pv = piv[l];
  const float v = (pv >= 0) ? dv[((size_t)b * 512 + pv) * 512 + c] : mtok[c];
  Xd[idx] = v + posd[(size_t)l * 512 + c];
}

// tok (8,1024,64) -> image (8,1,256,256), optional clip +-6
__global__ void unpatchify_kernel(const float* __restrict__ tok, float* __restrict__ out, int clip) {
  const int idx = blockIdx.x * 256 + threadIdx.x;   // 524,288
  const int iw = idx & 255;
  const int ih = (idx >> 8) & 255;
  const int b = idx >> 16;
  const int hp = ih >> 3, pr = ih & 7, wp = iw >> 3, pc = iw & 7;
  float v = tok[((size_t)b * 1024 + hp * 32 + wp) * 64 + pr * 8 + pc];
  if (clip) v = fminf(fmaxf(v, -6.0f), 6.0f);
  out[idx] = v;
}

__global__ void write_mask_kernel(float* __restrict__ out, const int* __restrict__ piv) {
  const int idx = blockIdx.x * 256 + threadIdx.x;   // 8192
  if (idx < 8192) out[idx] = (piv[idx & 1023] >= 0) ? 0.0f : 1.0f;
}

// ---------------------------------------------------------------------------
// Host orchestration
// ---------------------------------------------------------------------------
extern "C" void kernel_launch(void* const* d_in, const int* in_sizes, int n_in,
                              void* d_out, int out_size, void* d_ws, size_t ws_size,
                              hipStream_t stream)
{
  (void)in_sizes; (void)n_in; (void)out_size; (void)ws_size;
  const float* x       = (const float*)d_in[0];
  const int*   vis     = (const int*)d_in[1];
  const float* emb_w   = (const float*)d_in[2];
  const float* emb_b   = (const float*)d_in[3];
  const float* pos_enc = (const float*)d_in[4];
  const float* eln1s = (const float*)d_in[5];
  const float* eln1b = (const float*)d_in[6];
  const float* eqkvw = (const float*)d_in[7];
  const float* eqkvb = (const float*)d_in[8];
  const float* ebt   = (const float*)d_in[9];
  const float* epw   = (const float*)d_in[10];
  const float* epb   = (const float*)d_in[11];
  const float* eln2s = (const float*)d_in[12];
  const float* eln2b = (const float*)d_in[13];
  const float* ef1w  = (const float*)d_in[14];
  const float* ef1b  = (const float*)d_in[15];
  const float* ef2w  = (const float*)d_in[16];
  const float* ef2b  = (const float*)d_in[17];
  const float* dln1s = (const float*)d_in[18];
  const float* dln1b = (const float*)d_in[19];
  const float* dqkvw = (const float*)d_in[20];
  const float* dqkvb = (const float*)d_in[21];
  const float* dbt   = (const float*)d_in[22];
  const float* dpw   = (const float*)d_in[23];
  const float* dpb   = (const float*)d_in[24];
  const float* dln2s = (const float*)d_in[25];
  const float* dln2b = (const float*)d_in[26];
  const float* df1w  = (const float*)d_in[27];
  const float* df1b  = (const float*)d_in[28];
  const float* df2w  = (const float*)d_in[29];
  const float* df2b  = (const float*)d_in[30];
  const float* enorm_s = (const float*)d_in[31];
  const float* enorm_b = (const float*)d_in[32];
  const float* demb_w  = (const float*)d_in[33];
  const float* demb_b  = (const float*)d_in[34];
  const float* mask_tok = (const float*)d_in[35];
  const float* pos_dec  = (const float*)d_in[36];
  const float* dnorm_s = (const float*)d_in[37];
  const float* dnorm_b = (const float*)d_in[38];
  const float* muw = (const float*)d_in[39];
  const float* mub = (const float*)d_in[40];
  const float* lvw = (const float*)d_in[41];
  const float* lvb = (const float*)d_in[42];

  // arena over d_ws
  char* wsb = (char*)d_ws;
  size_t off = 0;
  auto alloc = [&](size_t bytes) -> void* {
    off = (off + 255) & ~(size_t)255;
    void* p = wsb + off;
    off += bytes;
    return p;
  };

  const int NR = 3969;  // (2*32-1)^2

  __bf16* w_emb  = (__bf16*)alloc((size_t)768 * 128 * 2);
  __bf16* w_eqkv = (__bf16*)alloc((size_t)6 * 2304 * 768 * 2);
  __bf16* w_epw  = (__bf16*)alloc((size_t)6 * 768 * 768 * 2);
  __bf16* w_ef1  = (__bf16*)alloc((size_t)6 * 3072 * 768 * 2);
  __bf16* w_ef2  = (__bf16*)alloc((size_t)6 * 768 * 3072 * 2);
  __bf16* w_dqkv = (__bf16*)alloc((size_t)4 * 1536 * 512 * 2);
  __bf16* w_dpw  = (__bf16*)alloc((size_t)4 * 512 * 512 * 2);
  __bf16* w_df1  = (__bf16*)alloc((size_t)4 * 2048 * 512 * 2);
  __bf16* w_df2  = (__bf16*)alloc((size_t)4 * 512 * 2048 * 2);
  __bf16* w_demb = (__bf16*)alloc((size_t)512 * 768 * 2);
  __bf16* w_mu   = (__bf16*)alloc((size_t)64 * 512 * 2);
  __bf16* w_lv   = (__bf16*)alloc((size_t)64 * 512 * 2);

  __bf16* tok_bf = (__bf16*)alloc((size_t)8 * 1024 * 128 * 2);
  __bf16* tv_bf  = (__bf16*)alloc((size_t)8 * 512 * 128 * 2);
  float*  Xe     = (float*)alloc((size_t)4096 * 768 * 4);
  __bf16* hb     = (__bf16*)alloc((size_t)4194304 * 2);     // max(4096*768, 8192*512)
  __bf16* qkv_bf = (__bf16*)alloc((size_t)12582912 * 2);    // max(4096*2304, 8192*1536)
  __bf16* O_bf   = (__bf16*)alloc((size_t)4194304 * 2);
  __bf16* hid_bf = (__bf16*)alloc((size_t)16777216 * 2);    // max(4096*3072, 8192*2048)
  float*  Xd     = (float*)alloc((size_t)8192 * 512 * 4);
  float*  dv     = (float*)alloc((size_t)4096 * 512 * 4);
  float*  mu_f   = (float*)alloc((size_t)8192 * 64 * 4);
  float*  lv_f   = (float*)alloc((size_t)8192 * 64 * 4);
  int*    piv    = (int*)alloc((size_t)1024 * 4);

  auto f2b = [&](const float* s, __bf16* d, long n) {
    f32_to_bf16_kernel<<<dim3((unsigned)((n + 255) / 256)), dim3(256), 0, stream>>>(s, d, n);
  };
  auto gemm = [&](const __bf16* X_, const __bf16* W_, const float* bias_,
                  const float* res_, float* Yf_, __bf16* Yb_,
                  int M_, int N_, int K_, int gelu_) {
    dim3 g((N_ + 127) / 128, M_ / 128);
    gemm_bf16_kernel<<<g, 256, 0, stream>>>(X_, W_, bias_, res_, Yf_, Yb_, M_, N_, K_, gelu_);
  };
  auto attn_smem = [](int Lx) -> size_t {
    return (size_t)16 * Lx * 4 + (size_t)16 * Lx * 2 + (size_t)Lx * 72 * 2 +
           (size_t)16 * 72 * 2 + 16 * 4 + 16 * 4;
  };
  (void)hipFuncSetAttribute(reinterpret_cast<const void*>(attn_kernel),
                            hipFuncAttributeMaxDynamicSharedMemorySize,
                            (int)attn_smem(1024));

  // weight conversion f32 -> bf16
  f2b(emb_w,  w_emb,  (long)768 * 128);
  f2b(eqkvw,  w_eqkv, (long)6 * 2304 * 768);
  f2b(epw,    w_epw,  (long)6 * 768 * 768);
  f2b(ef1w,   w_ef1,  (long)6 * 3072 * 768);
  f2b(ef2w,   w_ef2,  (long)6 * 768 * 3072);
  f2b(dqkvw,  w_dqkv, (long)4 * 1536 * 512);
  f2b(dpw,    w_dpw,  (long)4 * 512 * 512);
  f2b(df1w,   w_df1,  (long)4 * 2048 * 512);
  f2b(df2w,   w_df2,  (long)4 * 512 * 2048);
  f2b(demb_w, w_demb, (long)512 * 768);
  f2b(muw,    w_mu,   (long)64 * 512);
  f2b(lvw,    w_lv,   (long)64 * 512);

  // patchify + visible gather + embed (+pos)
  patchify_kernel<<<4096, 256, 0, stream>>>(x, tok_bf);
  gather_tokens_kernel<<<2048, 256, 0, stream>>>(tok_bf, vis, tv_bf);
  gemm(tv_bf, w_emb, emb_b, nullptr, Xe, nullptr, 4096, 768, 128, 0);
  add_pos_vis_kernel<<<12288, 256, 0, stream>>>(Xe, pos_enc, vis);

  // encoder blocks
  for (int d = 0; d < 6; ++d) {
    layernorm_bf16_kernel<<<4096, 256, 0, stream>>>(Xe, eln1s + d * 768, eln1b + d * 768, hb, 768);
    gemm(hb, w_eqkv + (size_t)d * 2304 * 768, eqkvb + d * 2304, nullptr, nullptr, qkv_bf,
         4096, 2304, 768, 0);
    attn_kernel<<<dim3(32, 12, 8), 256, attn_smem(512), stream>>>(
        qkv_bf, ebt + (size_t)d * NR * 12, vis, O_bf, 512, 768, 12);
    gemm(O_bf, w_epw + (size_t)d * 768 * 768, epb + d * 768, Xe, Xe, nullptr, 4096, 768, 768, 0);
    layernorm_bf16_kernel<<<4096, 256, 0, stream>>>(Xe, eln2s + d * 768, eln2b + d * 768, hb, 768);
    gemm(hb, w_ef1 + (size_t)d * 3072 * 768, ef1b + d * 3072, nullptr, nullptr, hid_bf,
         4096, 3072, 768, 1);
    gemm(hid_bf, w_ef2 + (size_t)d * 768 * 3072, ef2b + d * 768, Xe, Xe, nullptr,
         4096, 768, 3072, 0);
  }

  // encoder norm + decoder embed + scatter with mask token + pos
  layernorm_bf16_kernel<<<4096, 256, 0, stream>>>(Xe, enorm_s, enorm_b, hb, 768);
  gemm(hb, w_demb, demb_b, nullptr, dv, nullptr, 4096, 512, 768, 0);
  posinvis_init_kernel<<<4, 256, 0, stream>>>(piv);
  posinvis_set_kernel<<<2, 256, 0, stream>>>(vis, piv);
  scatter_dec_kernel<<<16384, 256, 0, stream>>>(Xd, dv, mask_tok, pos_dec, piv);

  // decoder blocks
  for (int d = 0; d < 4; ++d) {
    layernorm_bf16_kernel<<<8192, 256, 0, stream>>>(Xd, dln1s + d * 512, dln1b + d * 512, hb, 512);
    gemm(hb, w_dqkv + (size_t)d * 1536 * 512, dqkvb + d * 1536, nullptr, nullptr, qkv_bf,
         8192, 1536, 512, 0);
    attn_kernel<<<dim3(64, 8, 8), 256, attn_smem(1024), stream>>>(
        qkv_bf, dbt + (size_t)d * NR * 8, nullptr, O_bf, 1024, 512, 8);
    gemm(O_bf, w_dpw + (size_t)d * 512 * 512, dpb + d * 512, Xd, Xd, nullptr, 8192, 512, 512, 0);
    layernorm_bf16_kernel<<<8192, 256, 0, stream>>>(Xd, dln2s + d * 512, dln2b + d * 512, hb, 512);
    gemm(hb, w_df1 + (size_t)d * 2048 * 512, df1b + d * 2048, nullptr, nullptr, hid_bf,
         8192, 2048, 512, 1);
    gemm(hid_bf, w_df2 + (size_t)d * 512 * 2048, df2b + d * 512, Xd, Xd, nullptr,
         8192, 512, 2048, 0);
  }

  // heads + unpatchify + mask
  layernorm_bf16_kernel<<<8192, 256, 0, stream>>>(Xd, dnorm_s, dnorm_b, hb, 512);
  gemm(hb, w_mu, mub, nullptr, mu_f, nullptr, 8192, 64, 512, 0);
  gemm(hb, w_lv, lvb, nullptr, lv_f, nullptr, 8192, 64, 512, 0);

  float* out = (float*)d_out;
  unpatchify_kernel<<<2048, 256, 0, stream>>>(mu_f, out, 0);
  unpatchify_kernel<<<2048, 256, 0, stream>>>(lv_f, out + 524288, 1);
  write_mask_kernel<<<32, 256, 0, stream>>>(out + 1048576, piv);
}
